// Attention_16853451669997
// MI455X (gfx1250) — matrix operand order repaired and checked by reading, not compiled, not run on hardware
//
#include <hip/hip_runtime.h>
#include <hip/hip_bf16.h>

#define NH     16
#define DH     64
#define N_CTX  2048
#define BATCH  4
#define BM     128   // query rows per block (8 waves x 16)
#define BN     64    // keys per flash iteration
#define NEG_INF (-3.402823466e38f)

typedef __attribute__((ext_vector_type(16))) __bf16 v16bf;
typedef __attribute__((ext_vector_type(8)))  __bf16 v8bf;
typedef __attribute__((ext_vector_type(8)))  float  v8f;
typedef int v4i __attribute__((vector_size(16)));

#if __has_builtin(__builtin_amdgcn_global_load_async_to_lds_b128)
#define ASYNC_LDS 1
#else
#define ASYNC_LDS 0
#endif

#define AS1 __attribute__((address_space(1)))
#define AS3 __attribute__((address_space(3)))

__device__ __forceinline__ void async_copy16(const void* g, void* l) {
#if ASYNC_LDS
  __builtin_amdgcn_global_load_async_to_lds_b128((AS1 v4i*)g, (AS3 v4i*)l, 0, 0);
#else
  *(uint4*)l = *(const uint4*)g;
#endif
}

__device__ __forceinline__ void wait_async_le4() {
#if ASYNC_LDS
#if __has_builtin(__builtin_amdgcn_s_wait_asynccnt)
  __builtin_amdgcn_s_wait_asynccnt(4);
#else
  asm volatile("s_wait_asynccnt 0x4" ::: "memory");
#endif
#endif
}

__device__ __forceinline__ void wait_async_le0() {
#if ASYNC_LDS
#if __has_builtin(__builtin_amdgcn_s_wait_asynccnt)
  __builtin_amdgcn_s_wait_asynccnt(0);
#else
  asm volatile("s_wait_asynccnt 0x0" ::: "memory");
#endif
#endif
}

__device__ __forceinline__ v16bf cat16(v8bf lo, v8bf hi) {
  v16bf r;
#pragma unroll
  for (int i = 0; i < 8; ++i) { r[i] = lo[i]; r[i + 8] = hi[i]; }
  return r;
}

// load 8 consecutive f32, convert to bf16
__device__ __forceinline__ v8bf cvt8(const float* __restrict__ p) {
  v8bf r;
#pragma unroll
  for (int i = 0; i < 8; ++i) r[i] = (__bf16)p[i];
  return r;
}

__device__ __forceinline__ void wave_lds_fence() {
#if __has_builtin(__builtin_amdgcn_s_wait_dscnt)
  __builtin_amdgcn_s_wait_dscnt(0);
#else
  asm volatile("s_wait_dscnt 0x0" ::: "memory");
#endif
  __builtin_amdgcn_wave_barrier();
}

// ---------------------------------------------------------------------------
// Kernel 1: per-head projections q = xWq^T, k = xWk^T, v = xWv^T (bf16 out).
// q,k stored row-major [bh][n][64]; v stored transposed [bh][64][n].
// One wave handles one 16-row tile for one (b,h).
// ---------------------------------------------------------------------------
__global__ __launch_bounds__(256) void proj_kernel(
    const float* __restrict__ Qv, const float* __restrict__ Kv,
    const float* __restrict__ Vv,
    const float* __restrict__ WQ, const float* __restrict__ WK,
    const float* __restrict__ WV,
    __bf16* __restrict__ q, __bf16* __restrict__ k, __bf16* __restrict__ vT)
{
  const int lane   = threadIdx.x & 31;
  const int wave   = threadIdx.x >> 5;
  const int tile   = blockIdx.x * 8 + wave;        // 8192 tiles total
  const int bh     = tile >> 7;                    // tile / 128
  const int tn     = tile & 127;
  const int b      = bh >> 4;
  const int h      = bh & 15;
  const int row0   = tn * 16;
  const int ln15   = lane & 15;
  const int hiHalf = lane >> 4;                    // 0 | 1

  const int rowA = row0 + ln15;                    // A-fragment row (both halves)
  const float* srcs[3] = {Qv, Kv, Vv};
  const float* wts[3]  = {WQ, WK, WV};

#pragma unroll
  for (int mtx = 0; mtx < 3; ++mtx) {
    const float* X = srcs[mtx] + ((size_t)(b * N_CTX + rowA)) * (NH * DH) + h * DH;

    // A fragments: lanes 0-15 K={0..7,16..23}+c*32 ; lanes 16-31 K={8..15,24..31}+c*32
    v16bf afrag[2];
#pragma unroll
    for (int c = 0; c < 2; ++c) {
      const int koff = c * 32 + hiHalf * 8;
      afrag[c] = cat16(cvt8(X + koff), cvt8(X + koff + 16));
    }

    const float* W = wts[mtx];
    v8f acc[4] = {};
#pragma unroll
    for (int t = 0; t < 4; ++t) {
      const int ocol = t * 16 + ln15;              // B lane = output column
#pragma unroll
      for (int c = 0; c < 2; ++c) {
        const int kbase = c * 32 + hiHalf * 8;     // B: lanes0-15 K={0..7,16..23}, 16-31 K={8..15,24..31}
        const float* wp = W + (size_t)ocol * DH + kbase;
        v16bf bfrag = cat16(cvt8(wp), cvt8(wp + 16));
        acc[t] = __builtin_amdgcn_wmma_f32_16x16x32_bf16(
            false, afrag[c], false, bfrag, (short)0, acc[t], false, false);
      }
    }

    // C/D layout: lane = N(col), VGPR r = row hiHalf*8 + r
    const size_t base_qk = (size_t)bh * N_CTX * DH;
#pragma unroll
    for (int t = 0; t < 4; ++t) {
      const int col = t * 16 + ln15;
#pragma unroll
      for (int r = 0; r < 8; ++r) {
        const int row = row0 + hiHalf * 8 + r;
        __bf16 val = (__bf16)acc[t][r];
        if (mtx == 0)      q[base_qk + (size_t)row * DH + col] = val;
        else if (mtx == 1) k[base_qk + (size_t)row * DH + col] = val;
        else               vT[((size_t)bh * DH + col) * N_CTX + row] = val;
      }
    }
  }
}

// ---------------------------------------------------------------------------
// Kernel 2: flash attention. 8 waves/block, wave owns 16 query rows.
// K and V^T tiles (64 keys) double-buffered in LDS via async-to-LDS DMA.
// ---------------------------------------------------------------------------
__global__ __launch_bounds__(256) void fa_kernel(
    const __bf16* __restrict__ q, const __bf16* __restrict__ k,
    const __bf16* __restrict__ vT, const unsigned char* __restrict__ mask,
    float* __restrict__ out)
{
  __shared__ __align__(16) __bf16 ktile[2][BN][DH];   // [buf][key][dim] 16KB
  __shared__ __align__(16) __bf16 vtile[2][DH][BN];   // [buf][dim][key] 16KB
  __shared__ __align__(16) __bf16 pbuf[8][16][BN];    // per-wave P     16KB

  const int tid    = threadIdx.x;
  const int lane   = tid & 31;
  const int wave   = tid >> 5;
  const int bh     = blockIdx.x >> 4;
  const int rblk   = blockIdx.x & 15;
  const int b      = bh >> 4;
  const int h      = bh & 15;
  const int ln15   = lane & 15;
  const int hiHalf = lane >> 4;

  const int    row0   = rblk * BM + wave * 16;
  const size_t qkbase = (size_t)bh * N_CTX * DH;
  const size_t vbase  = (size_t)bh * DH * N_CTX;

  // stage one 64-key tile (K: 8KB contiguous; V^T: 64 rows strided by N_CTX).
  // Exactly 4 async-load instructions per thread -> static ASYNCcnt bookkeeping.
  auto stage = [&](int it, int buf) {
    const uint4* ks = (const uint4*)(k + qkbase + (size_t)it * BN * DH);
    uint4* kd = (uint4*)&ktile[buf][0][0];
    async_copy16(&ks[tid], &kd[tid]);
    async_copy16(&ks[tid + 256], &kd[tid + 256]);
    const __bf16* vs = vT + vbase + it * BN;
    {
      int i = tid, d = i >> 3, c = i & 7;
      async_copy16(vs + (size_t)d * N_CTX + c * 8, &vtile[buf][d][c * 8]);
      i = tid + 256; d = i >> 3; c = i & 7;
      async_copy16(vs + (size_t)d * N_CTX + c * 8, &vtile[buf][d][c * 8]);
    }
  };

  // Q A-fragments for this wave's rows (held for the whole pass)
  const __bf16* qrow = q + qkbase + (size_t)(row0 + ln15) * DH;
  v16bf qfrag[2];
#pragma unroll
  for (int c = 0; c < 2; ++c) {
    const int koff = c * 32 + hiHalf * 8;
    qfrag[c] = cat16(*(const v8bf*)(qrow + koff), *(const v8bf*)(qrow + koff + 16));
  }

  // row masks for this lane's 8 rows (rows hiHalf*8 + r)
  float rowmask[8];
#pragma unroll
  for (int r = 0; r < 8; ++r)
    rowmask[r] = mask[(size_t)b * N_CTX + row0 + hiHalf * 8 + r] ? 1.f : 0.f;

  float m_i[8], l_i[8];
  v8f   o_acc[4] = {};
#pragma unroll
  for (int r = 0; r < 8; ++r) { m_i[r] = NEG_INF; l_i[r] = 0.f; }

  const float scale = 0.125f;  // 1/sqrt(64)
  const int nIter = N_CTX / BN;

  stage(0, 0);  // prime buffer 0

  for (int it = 0; it < nIter; ++it) {
    const int p = it & 1;
    const int key0 = it * BN;

    if (it + 1 < nIter) {        // prefetch next tile into the other buffer
      stage(it + 1, p ^ 1);
      wait_async_le4();          // current tile's 4 loads (issued earlier) done
    } else {
      wait_async_le0();
    }
    __syncthreads();             // everyone's current tile landed

    // ---- S = Q K^T : 4 column tiles x 2 K-chunks ----
    v8f s[4] = {};
#pragma unroll
    for (int t = 0; t < 4; ++t) {
      const __bf16* kp = &ktile[p][t * 16 + ln15][0];   // B lane = key column
#pragma unroll
      for (int c = 0; c < 2; ++c) {
        const int kb = c * 32 + hiHalf * 8;
        v16bf bfrag = cat16(*(const v8bf*)(kp + kb), *(const v8bf*)(kp + kb + 16));
        s[t] = __builtin_amdgcn_wmma_f32_16x16x32_bf16(
            false, qfrag[c], false, bfrag, (short)0, s[t], false, false);
      }
    }

    // ---- scale + pair mask ----
    float colmask[4];
#pragma unroll
    for (int t = 0; t < 4; ++t)
      colmask[t] = mask[(size_t)b * N_CTX + key0 + t * 16 + ln15] ? 1.f : 0.f;

    float sm[4][8];
#pragma unroll
    for (int t = 0; t < 4; ++t)
#pragma unroll
      for (int r = 0; r < 8; ++r)
        sm[t][r] = (colmask[t] * rowmask[r] > 0.f) ? s[t][r] * scale : NEG_INF;

    // ---- online softmax: row max across tiles then 16-lane half-wave ----
    float rmax[8];
#pragma unroll
    for (int r = 0; r < 8; ++r) {
      float v = fmaxf(fmaxf(sm[0][r], sm[1][r]), fmaxf(sm[2][r], sm[3][r]));
#pragma unroll
      for (int off = 1; off < 16; off <<= 1)
        v = fmaxf(v, __shfl_xor(v, off, 32));
      rmax[r] = v;
    }

    float corr[8];
#pragma unroll
    for (int r = 0; r < 8; ++r) {
      const float mn = fmaxf(m_i[r], rmax[r]);
      corr[r] = __expf(m_i[r] - mn);
      m_i[r]  = mn;
    }

    float rsum[8] = {0, 0, 0, 0, 0, 0, 0, 0};
#pragma unroll
    for (int t = 0; t < 4; ++t)
#pragma unroll
      for (int r = 0; r < 8; ++r) {
        const float pex = __expf(sm[t][r] - m_i[r]);
        rsum[r] += pex;
        pbuf[wave][hiHalf * 8 + r][t * 16 + ln15] = (__bf16)pex;  // C/D -> LDS
      }

#pragma unroll
    for (int r = 0; r < 8; ++r) {
      float v = rsum[r];
#pragma unroll
      for (int off = 1; off < 16; off <<= 1)
        v += __shfl_xor(v, off, 32);
      l_i[r] = l_i[r] * corr[r] + v;
    }
#pragma unroll
    for (int t = 0; t < 4; ++t)
#pragma unroll
      for (int r = 0; r < 8; ++r)
        o_acc[t][r] *= corr[r];

    wave_lds_fence();  // P stores visible before A-fragment re-read (wave-local)

    // ---- O += P @ V : re-read P in A layout, V^T rows give B fragments ----
#pragma unroll
    for (int c = 0; c < 2; ++c) {
      const int koff = c * 32 + hiHalf * 8;
      const __bf16* pr = &pbuf[wave][ln15][0];
      v16bf pfrag = cat16(*(const v8bf*)(pr + koff), *(const v8bf*)(pr + koff + 16));
#pragma unroll
      for (int t = 0; t < 4; ++t) {
        const __bf16* vp = &vtile[p][t * 16 + ln15][0];  // B lane = output dim
        const int kb = c * 32 + hiHalf * 8;
        v16bf bfrag = cat16(*(const v8bf*)(vp + kb), *(const v8bf*)(vp + kb + 16));
        o_acc[t] = __builtin_amdgcn_wmma_f32_16x16x32_bf16(
            false, pfrag, false, bfrag, (short)0, o_acc[t], false, false);
      }
    }

    __syncthreads();  // all waves done with buffer p before it is re-staged
  }

  // ---- epilogue: normalize and write out[b, row, h*64 + dim] ----
#pragma unroll
  for (int t = 0; t < 4; ++t) {
    const int dim = t * 16 + ln15;
#pragma unroll
    for (int r = 0; r < 8; ++r) {
      const int row = row0 + hiHalf * 8 + r;
      out[((size_t)(b * N_CTX + row)) * (NH * DH) + h * DH + dim] =
          o_acc[t][r] / l_i[r];
    }
  }
}

extern "C" void kernel_launch(void* const* d_in, const int* in_sizes, int n_in,
                              void* d_out, int out_size, void* d_ws, size_t ws_size,
                              hipStream_t stream) {
  (void)in_sizes; (void)n_in; (void)out_size; (void)ws_size;
  const float* Qv = (const float*)d_in[0];
  const float* Kv = (const float*)d_in[1];
  const float* Vv = (const float*)d_in[2];
  const float* WQ = (const float*)d_in[3];
  const float* WK = (const float*)d_in[4];
  const float* WV = (const float*)d_in[5];
  const unsigned char* mask = (const unsigned char*)d_in[6];
  float* out = (float*)d_out;

  const size_t per = (size_t)BATCH * NH * N_CTX * DH;  // 8.39M elems (16 MB bf16)
  __bf16* q  = (__bf16*)d_ws;
  __bf16* k  = q + per;
  __bf16* vT = k + per;

  // 8192 row-tiles / 8 waves per block
  proj_kernel<<<1024, 256, 0, stream>>>(Qv, Kv, Vv, WQ, WK, WV, q, k, vT);
  // 64 (b,h) x 16 row-blocks
  fa_kernel<<<1024, 256, 0, stream>>>(q, k, vT, mask, out);
}
